// TransformerLayer_18322330484747
// MI455X (gfx1250) — matrix operand order repaired and checked by reading, not compiled, not run on hardware
//
#include <hip/hip_runtime.h>
#include <cstdint>
#include <cstddef>

// ---------------------------------------------------------------------------
// MI455X / gfx1250 implementation. All matmuls (projections, FFNs, attention
// score & context products) go through v_wmma_f32_16x16x32_f16 (wave32).
// Flash-style attention (online softmax); activations staged as f16 with f32
// WMMA accumulation. All WMMA fragments load as contiguous ds_load_b128 pairs;
// global->LDS tile staging uses CDNA5 async copies (ASYNCcnt) when available.
// ---------------------------------------------------------------------------

typedef __attribute__((ext_vector_type(16))) _Float16 v16h;
typedef __attribute__((ext_vector_type(8)))  float    v8f;

constexpr int BB   = 2;
constexpr int NN   = 2048;
constexpr int DD   = 256;
constexpr int HH   = 4;
constexpr int HDIM = 64;
constexpr int BN   = BB * NN;   // 4096 rows for every GEMM
constexpr int D2   = 2 * DD;    // 512

union V16U {            // assemble a 16-half WMMA operand from two 16B loads
  uint4 q[2];
  v16h  h;
};

__device__ __forceinline__ v8f wmma16(v16h a, v16h b, v8f c) {
  return __builtin_amdgcn_wmma_f32_16x16x32_f16(false, a, false, b, (short)0, c,
                                                false, false);
}

// ---- CDNA5 async global->LDS copy (GLOBAL_LOAD_ASYNC_TO_LDS_B128) ----------
#if defined(__has_builtin)
#if __has_builtin(__builtin_amdgcn_global_load_async_to_lds_b128)
#define HAVE_ASYNC_COPY 1
#endif
#endif
#ifndef HAVE_ASYNC_COPY
#define HAVE_ASYNC_COPY 0
#endif

typedef int v4i __attribute__((vector_size(4 * sizeof(int))));
typedef __attribute__((address_space(1))) v4i gv4i;   // global (AS1)
typedef __attribute__((address_space(3))) v4i lv4i;   // LDS (AS3)

__device__ __forceinline__ void cp_b128(void* lds, const void* g) {
#if HAVE_ASYNC_COPY
  __builtin_amdgcn_global_load_async_to_lds_b128((gv4i*)g, (lv4i*)lds, 0, 0);
#else
  *(uint4*)lds = *(const uint4*)g;
#endif
}
__device__ __forceinline__ void cp_wait() {
#if HAVE_ASYNC_COPY
#if defined(__has_builtin) && __has_builtin(__builtin_amdgcn_s_wait_asynccnt)
  __builtin_amdgcn_s_wait_asynccnt(0);
#else
  asm volatile("s_wait_asynccnt 0x0" ::: "memory");
#endif
#endif
}

// ---------------------------------------------------------------------------
// Generic WMMA GEMM: C[M,Nn] = A[M,K] (f16) @ W (f16, PRE-TRANSPOSED [Nn,K])
//                    + bias (f32)
// EPI 0: f16 out at C16[row*ldc + coff + col]
// EPI 1: f32 out at C32[row*ldc + coff + col]
// EPI 2: f16 out, += residual R (f16, [M,Nn])
// EPI 3: f32 out, += residual R (final model outputs into d_out)
// EPI 4: f16 transposed store: C16[(b*DD + col)*NN + n]  (d-major V producer;
//        requires M == BN, Nn == DD)
// Block: 128 threads (4 waves), tile 64x64, K-step 32.
// ---------------------------------------------------------------------------
template <int EPI>
__global__ __launch_bounds__(128) void gemm_wmma(
    const _Float16* __restrict__ A, const _Float16* __restrict__ Wt,
    const float* __restrict__ bias, const _Float16* __restrict__ R,
    _Float16* __restrict__ C16, float* __restrict__ C32,
    int M, int Nn, int K, int ldc, int coff) {
  __shared__ _Float16 As[64][32 + 8];   // [m][k]
  __shared__ _Float16 Bs[64][32 + 8];   // [n][k]  (weight tile, k contiguous)
  const int tid = threadIdx.x;
  const int wv = tid >> 5, lane = tid & 31;
  const int m16 = lane & 15, khf = lane >> 4;
  const int mbase = blockIdx.x * 64;
  const int nbase = blockIdx.y * 64;

  // two copy slots per thread, addresses hoisted out of the K loop
  const int r0 = tid >> 2, c0 = (tid & 3) * 8;   // slot 0: rows 0..31
  const int r1 = r0 + 32;                        // slot 1: rows 32..63
  const _Float16* pA0 = A + (size_t)(mbase + r0) * K + c0;
  const _Float16* pA1 = A + (size_t)(mbase + r1) * K + c0;
  const _Float16* pB0 = Wt + (size_t)(nbase + r0) * K + c0;
  const _Float16* pB1 = Wt + (size_t)(nbase + r1) * K + c0;
  _Float16* sA0 = &As[r0][c0];
  _Float16* sA1 = &As[r1][c0];
  _Float16* sB0 = &Bs[r0][c0];
  _Float16* sB1 = &Bs[r1][c0];

  v8f acc[4] = {};
  for (int k0 = 0; k0 < K;
       k0 += 32, pA0 += 32, pA1 += 32, pB0 += 32, pB1 += 32) {
    __syncthreads();
    cp_b128(sA0, pA0);
    cp_b128(sA1, pA1);
    cp_b128(sB0, pB0);
    cp_b128(sB1, pB1);
    cp_wait();
    __syncthreads();
    V16U au;                                     // A frag: 2 contiguous b128
    au.q[0] = *(uint4*)&As[wv * 16 + m16][khf * 8];
    au.q[1] = *(uint4*)&As[wv * 16 + m16][16 + khf * 8];
#pragma unroll
    for (int c = 0; c < 4; ++c) {                // B frag: 2 contiguous b128
      V16U bu;
      bu.q[0] = *(uint4*)&Bs[c * 16 + m16][khf * 8];
      bu.q[1] = *(uint4*)&Bs[c * 16 + m16][16 + khf * 8];
      acc[c] = wmma16(au.h, bu.h, acc[c]);
    }
  }
#pragma unroll
  for (int c = 0; c < 4; ++c) {
#pragma unroll
    for (int r = 0; r < 8; ++r) {
      int row = mbase + wv * 16 + khf * 8 + r;
      int col = nbase + c * 16 + m16;
      float v = acc[c][r] + bias[col];
      if (EPI == 2 || EPI == 3) v += (float)R[(size_t)row * Nn + col];
      if (EPI == 0 || EPI == 2) {
        C16[(size_t)row * ldc + coff + col] = (_Float16)v;
      } else if (EPI == 4) {
        int brow = row >> 11, nrow = row & (NN - 1);
        C16[((size_t)brow * DD + col) * (size_t)NN + nrow] = (_Float16)v;
      } else {
        C32[(size_t)row * ldc + coff + col] = v;
      }
    }
  }
}

// ---------------------------------------------------------------------------
// Flash attention (online softmax). Grid: (NN/64, HH, BB), block 128 (4 waves).
// BHND=true : Q/K laid out [B,H,N,HD] (self-attn after RoPE)
// BHND=false: Q/K laid out [B,N,H,HD] (cross-attn head views)
// V is ALWAYS d-major: V[(b*DD + h*HD + d)*NN + n].
// Output ctx written [B,N,H*HD] f16.
// ---------------------------------------------------------------------------
template <bool BHND>
__global__ __launch_bounds__(128) void flash_attn(
    const _Float16* __restrict__ Qp, const _Float16* __restrict__ Kp,
    const _Float16* __restrict__ Vt, _Float16* __restrict__ Op, float scale) {
  __shared__ _Float16 Ks[32][HDIM + 8];   // [key][d]
  __shared__ _Float16 Vs[64][32 + 8];     // [d][key]
  __shared__ _Float16 Ps[4][16][40];      // per-wave P tile [m][key]
  const int tid = threadIdx.x, wv = tid >> 5, lane = tid & 31;
  const int m16 = lane & 15, khf = lane >> 4;
  const int b = blockIdx.z, h = blockIdx.y;
  const int q0 = blockIdx.x * 64 + wv * 16;

  auto gidx = [&](int n, int d) -> size_t {
    if (BHND) return ((size_t)(b * HH + h) * NN + n) * HDIM + d;
    return ((size_t)(b * NN + n)) * DD + h * HDIM + d;
  };

  V16U qf[2];
#pragma unroll
  for (int dc = 0; dc < 2; ++dc) {
    qf[dc].q[0] = *(const uint4*)&Qp[gidx(q0 + m16, dc * 32 + khf * 8)];
    qf[dc].q[1] = *(const uint4*)&Qp[gidx(q0 + m16, dc * 32 + 16 + khf * 8)];
  }

  // tile-copy slots (hoisted): K tile 32x64 halves, V tile 64x32 halves
  const size_t krowstr = BHND ? HDIM : DD;              // halves per key row
  const int kr0 = tid >> 3, kc0 = (tid & 7) * 8;        // K rows 0..15
  const int kr1 = kr0 + 16;                             // K rows 16..31
  const _Float16* pK0 = Kp + gidx(kr0, kc0);
  const _Float16* pK1 = Kp + gidx(kr1, kc0);
  _Float16* sK0 = &Ks[kr0][kc0];
  _Float16* sK1 = &Ks[kr1][kc0];
  const int vr0 = tid >> 2, vc0 = (tid & 3) * 8;        // V d-rows 0..31
  const int vr1 = vr0 + 32;                             // V d-rows 32..63
  const _Float16* pV0 =
      Vt + ((size_t)(b * DD + h * HDIM + vr0)) * NN + vc0;
  const _Float16* pV1 =
      Vt + ((size_t)(b * DD + h * HDIM + vr1)) * NN + vc0;
  _Float16* sV0 = &Vs[vr0][vc0];
  _Float16* sV1 = &Vs[vr1][vc0];

  v8f o[4] = {};
  float mrow[8], lrow[8];
#pragma unroll
  for (int r = 0; r < 8; ++r) { mrow[r] = -1e30f; lrow[r] = 0.f; }

  for (int kt = 0; kt < NN;
       kt += 32, pK0 += 32 * krowstr, pK1 += 32 * krowstr, pV0 += 32,
       pV1 += 32) {
    __syncthreads();
    cp_b128(sK0, pK0);
    cp_b128(sK1, pK1);
    cp_b128(sV0, pV0);
    cp_b128(sV1, pV1);
    cp_wait();
    __syncthreads();

    v8f s[2];
#pragma unroll
    for (int cj = 0; cj < 2; ++cj) {
      v8f t = {};
#pragma unroll
      for (int dc = 0; dc < 2; ++dc) {           // B = K^T: d contiguous
        V16U ku;
        ku.q[0] = *(uint4*)&Ks[cj * 16 + m16][dc * 32 + khf * 8];
        ku.q[1] = *(uint4*)&Ks[cj * 16 + m16][dc * 32 + 16 + khf * 8];
        t = wmma16(qf[dc].h, ku.h, t);
      }
      s[cj] = t;
    }
#pragma unroll
    for (int r = 0; r < 8; ++r) {
      float s0 = s[0][r] * scale, s1 = s[1][r] * scale;
      float mx = fmaxf(s0, s1);
#pragma unroll
      for (int d = 1; d < 16; d <<= 1) mx = fmaxf(mx, __shfl_xor(mx, d, 32));
      float nm = fmaxf(mrow[r], mx);
      float alpha = __expf(mrow[r] - nm);
      mrow[r] = nm;
      float p0 = __expf(s0 - nm), p1 = __expf(s1 - nm);
      float rs = p0 + p1;
#pragma unroll
      for (int d = 1; d < 16; d <<= 1) rs += __shfl_xor(rs, d, 32);
      lrow[r] = lrow[r] * alpha + rs;
      Ps[wv][khf * 8 + r][m16] = (_Float16)p0;
      Ps[wv][khf * 8 + r][16 + m16] = (_Float16)p1;
#pragma unroll
      for (int c = 0; c < 4; ++c) o[c][r] = o[c][r] * alpha;
    }
    asm volatile("s_wait_dscnt 0x0" ::: "memory");  // wave-local LDS RAW
    V16U pu;
    pu.q[0] = *(uint4*)&Ps[wv][m16][khf * 8];
    pu.q[1] = *(uint4*)&Ps[wv][m16][16 + khf * 8];
#pragma unroll
    for (int c = 0; c < 4; ++c) {                // B = V: key contiguous
      V16U vu;
      vu.q[0] = *(uint4*)&Vs[c * 16 + m16][khf * 8];
      vu.q[1] = *(uint4*)&Vs[c * 16 + m16][16 + khf * 8];
      o[c] = wmma16(pu.h, vu.h, o[c]);
    }
  }
#pragma unroll
  for (int r = 0; r < 8; ++r) {
    float inv = 1.0f / lrow[r];
    int row = q0 + khf * 8 + r;
#pragma unroll
    for (int c = 0; c < 4; ++c)
      Op[((size_t)(b * NN + row)) * DD + h * HDIM + c * 16 + m16] =
          (_Float16)(o[c][r] * inv);
  }
}

// ---------------------------------------------------------------------------
// Adjacency attention: ctx = softmax(adj or adj^T, rows) @ V.
// V is d-major: V[(b*DD + d)*NN + n]. Grid: (NN/64, DD/64, BB), block 128.
// ---------------------------------------------------------------------------
template <bool TRANSP>
__global__ __launch_bounds__(128) void flash_adj(
    const float* __restrict__ adj, const _Float16* __restrict__ Vt,
    _Float16* __restrict__ Op) {
  __shared__ _Float16 Vs[64][32 + 8];   // [d][key]
  __shared__ _Float16 Ps[4][16][40];
  const int tid = threadIdx.x, wv = tid >> 5, lane = tid & 31;
  const int m16 = lane & 15, khf = lane >> 4;
  const int b = blockIdx.z, d0 = blockIdx.y * 64;
  const int row0 = blockIdx.x * 64 + wv * 16;
  const float* adjb = adj + (size_t)b * NN * NN;

  // hoisted score pointer: advances by 32 (rows-softmax) or 32*NN (cols)
  const float* ps = TRANSP ? adjb + (size_t)m16 * NN + row0 + khf * 8
                           : adjb + (size_t)(row0 + khf * 8) * NN + m16;
  const size_t pstep = TRANSP ? (size_t)32 * NN : 32;

  const int vr0 = tid >> 2, vc0 = (tid & 3) * 8;
  const int vr1 = vr0 + 32;
  const _Float16* pV0 = Vt + ((size_t)(b * DD + d0 + vr0)) * NN + vc0;
  const _Float16* pV1 = Vt + ((size_t)(b * DD + d0 + vr1)) * NN + vc0;
  _Float16* sV0 = &Vs[vr0][vc0];
  _Float16* sV1 = &Vs[vr1][vc0];

  v8f o[4] = {};
  float mrow[8], lrow[8];
#pragma unroll
  for (int r = 0; r < 8; ++r) { mrow[r] = -1e30f; lrow[r] = 0.f; }

  for (int kt = 0; kt < NN; kt += 32, ps += pstep, pV0 += 32, pV1 += 32) {
    __syncthreads();
    cp_b128(sV0, pV0);
    cp_b128(sV1, pV1);
    cp_wait();
    __syncthreads();
#pragma unroll
    for (int r = 0; r < 8; ++r) {
      float s0 = TRANSP ? ps[r] : ps[(size_t)r * NN];
      float s1 = TRANSP ? ps[(size_t)16 * NN + r] : ps[(size_t)r * NN + 16];
      float mx = fmaxf(s0, s1);
#pragma unroll
      for (int d = 1; d < 16; d <<= 1) mx = fmaxf(mx, __shfl_xor(mx, d, 32));
      float nm = fmaxf(mrow[r], mx);
      float alpha = __expf(mrow[r] - nm);
      mrow[r] = nm;
      float p0 = __expf(s0 - nm), p1 = __expf(s1 - nm);
      float rs = p0 + p1;
#pragma unroll
      for (int d = 1; d < 16; d <<= 1) rs += __shfl_xor(rs, d, 32);
      lrow[r] = lrow[r] * alpha + rs;
      Ps[wv][khf * 8 + r][m16] = (_Float16)p0;
      Ps[wv][khf * 8 + r][16 + m16] = (_Float16)p1;
#pragma unroll
      for (int c = 0; c < 4; ++c) o[c][r] = o[c][r] * alpha;
    }
    asm volatile("s_wait_dscnt 0x0" ::: "memory");
    V16U pu;
    pu.q[0] = *(uint4*)&Ps[wv][m16][khf * 8];
    pu.q[1] = *(uint4*)&Ps[wv][m16][16 + khf * 8];
#pragma unroll
    for (int c = 0; c < 4; ++c) {
      V16U vu;
      vu.q[0] = *(uint4*)&Vs[c * 16 + m16][khf * 8];
      vu.q[1] = *(uint4*)&Vs[c * 16 + m16][16 + khf * 8];
      o[c] = wmma16(pu.h, vu.h, o[c]);
    }
  }
#pragma unroll
  for (int r = 0; r < 8; ++r) {
    float inv = 1.0f / lrow[r];
    int row = row0 + khf * 8 + r;
#pragma unroll
    for (int c = 0; c < 4; ++c)
      Op[((size_t)(b * NN + row)) * DD + d0 + c * 16 + m16] =
          (_Float16)(o[c][r] * inv);
  }
}

// ---------------------------------------------------------------------------
// QKV split (unflatten(-1,(H,HD,3)) layout) + RoPE. One thread per (b,h,n,d).
// Q,K written [B,H,N,HD]; V written d-major [(b*H+h)*HD + d][n].
// ---------------------------------------------------------------------------
__global__ void qkv_rope(const _Float16* __restrict__ qkv,
                         const float* __restrict__ enc,
                         _Float16* __restrict__ Qh, _Float16* __restrict__ Kh,
                         _Float16* __restrict__ Vt) {
  size_t idx = (size_t)blockIdx.x * 256 + threadIdx.x;  // B*H*N*HD total
  int d = idx & 63;
  size_t t = idx >> 6;
  int n = t & (NN - 1);
  t >>= 11;
  int h = t & 3;
  int b = (int)(t >> 2);
  size_t base = ((size_t)(b * NN + n)) * (3 * DD) + h * (HDIM * 3);
  float q = (float)qkv[base + d * 3 + 0];
  float k = (float)qkv[base + d * 3 + 1];
  float v = (float)qkv[base + d * 3 + 2];
  int dp = d ^ 1;
  float qp = (float)qkv[base + dp * 3 + 0];
  float kp = (float)qkv[base + dp * 3 + 1];
  float rq = (d & 1) ? qp : -qp;  // rotate_half
  float rk = (d & 1) ? kp : -kp;
  size_t eb = ((size_t)b * NN + n) * HDIM + d;
  float f0 = enc[eb];
  float f1 = enc[(size_t)BB * NN * HDIM + eb];
  size_t oidx = ((size_t)(b * HH + h) * NN + n) * HDIM + d;
  Qh[oidx] = (_Float16)(q * f0 + rq * f1);
  Kh[oidx] = (_Float16)(k * f0 + rk * f1);
  Vt[((size_t)(b * HH + h) * HDIM + d) * NN + n] = (_Float16)v;
}

// ---------------------------------------------------------------------------
// Row LayerNorm (512-wide) + exact GELU. One wave per row (8 rows/block).
// ---------------------------------------------------------------------------
__global__ __launch_bounds__(256) void ln_gelu(const float* __restrict__ X,
                                               const float* __restrict__ g,
                                               const float* __restrict__ bg,
                                               _Float16* __restrict__ Y) {
  int wv = threadIdx.x >> 5, lane = threadIdx.x & 31;
  size_t row = (size_t)blockIdx.x * 8 + wv;
  const float* x = X + row * D2;
  float v[16], s = 0.f, ss = 0.f;
#pragma unroll
  for (int i = 0; i < 16; ++i) {
    v[i] = x[lane + i * 32];
    s += v[i];
    ss += v[i] * v[i];
  }
#pragma unroll
  for (int d = 1; d < 32; d <<= 1) {
    s += __shfl_xor(s, d, 32);
    ss += __shfl_xor(ss, d, 32);
  }
  float mean = s * (1.f / D2);
  float var = ss * (1.f / D2) - mean * mean;
  float rstd = rsqrtf(var + 1e-5f);
#pragma unroll
  for (int i = 0; i < 16; ++i) {
    int c = lane + i * 32;
    float y = (v[i] - mean) * rstd * g[c] + bg[c];
    Y[row * D2 + c] = (_Float16)(0.5f * y * (1.0f + erff(y * 0.7071067811865475f)));
  }
}

__global__ void cvt_f32_f16(const float* __restrict__ x,
                            _Float16* __restrict__ y, int n) {
  int i = blockIdx.x * 256 + threadIdx.x;
  if (i < n) y[i] = (_Float16)x[i];
}

// f32 [K,Nn] -> f16 transposed [Nn,K] (weight prep for the GEMM B side)
__global__ void cvt_f32_f16_t(const float* __restrict__ x,
                              _Float16* __restrict__ y, int K, int Nn) {
  int i = blockIdx.x * 256 + threadIdx.x;
  if (i < K * Nn) {
    int k = i / Nn, n = i % Nn;
    y[(size_t)n * K + k] = (_Float16)x[i];
  }
}

// strided f16 copy (used for concat packing); cols fixed per launch
__global__ void copy16(const _Float16* __restrict__ src, int sld,
                       _Float16* __restrict__ dst, int dld, int cols) {
  size_t i = (size_t)blockIdx.x * 256 + threadIdx.x;
  size_t row = i / cols;
  int c = (int)(i % cols);
  dst[row * dld + c] = src[row * sld + c];
}

// ce = relu(embd) @ W_embd + b_embd  (2 x 256 outputs, f32)
__global__ void ce_kernel(const float* __restrict__ embd,
                          const float* __restrict__ w,
                          const float* __restrict__ bias,
                          float* __restrict__ ce) {
  int j = blockIdx.x * 256 + threadIdx.x;  // 512 total
  int b = j >> 8, col = j & 255;
  float acc = bias[col];
  for (int k = 0; k < 128; ++k) {
    float e = embd[b * 128 + k];
    e = e > 0.f ? e : 0.f;
    acc += e * w[k * 256 + col];
  }
  ce[j] = acc;
}

// y = x + ce[batch]  (broadcast over n), f16 out
__global__ void add_ce_k(const _Float16* __restrict__ x,
                         const float* __restrict__ ce,
                         _Float16* __restrict__ y) {
  size_t i = (size_t)blockIdx.x * 256 + threadIdx.x;  // BN*DD
  int c = (int)(i & 255);
  size_t row = i >> 8;
  int b = (int)(row >> 11);
  y[i] = (_Float16)((float)x[i] + ce[b * 256 + c]);
}

// ---------------------------------------------------------------------------
// Host orchestration.
// Input flattening assumption (recursive dict-insertion order):
//  0 desc0  1 desc1  2 encoding0  3 encoding1  4 embd  5 adj_mat
//  6 embd.w 7 embd.b
//  8 wqkv 9 bqkv 10 wo_s 11 bo_s 12 w1_s 13 b1_s 14 g_s 15 bg_s 16 w2_s 17 b2_s
// 18 wv_a 19 bv_a 20 wo_a 21 bo_a 22 w1_a 23 b1_a 24 g_a 25 bg_a 26 w2_a 27 b2_a
// 28 wqk_c 29 bqk_c 30 wv_c 31 bv_c 32 wo_c 33 bo_c 34 w1_c 35 b1_c 36 g_c
// 37 bg_c 38 w2_c 39 b2_c 40 w_cp 41 b_cp
// ---------------------------------------------------------------------------
extern "C" void kernel_launch(void* const* d_in, const int* in_sizes, int n_in,
                              void* d_out, int out_size, void* d_ws,
                              size_t ws_size, hipStream_t stream) {
  (void)in_sizes; (void)n_in; (void)out_size; (void)ws_size;
  auto F = [&](int i) { return (const float*)d_in[i]; };

  const float* desc0 = F(0);
  const float* desc1 = F(1);
  const float* enc0 = F(2);
  const float* enc1 = F(3);
  const float* embd = F(4);
  const float* adj = F(5);

  // workspace bump allocator (256B aligned)
  char* base = (char*)d_ws;
  size_t off = 0;
  auto alloc = [&](size_t bytes) -> void* {
    void* p = base + off;
    off = (off + bytes + 255) & ~(size_t)255;
    return p;
  };
  auto ah = [&](size_t n) { return (_Float16*)alloc(n * sizeof(_Float16)); };
  auto af = [&](size_t n) { return (float*)alloc(n * sizeof(float)); };

  // f16 weights (all stored transposed: [Nn,K])
  _Float16 *Wqkv = ah(256 * 768), *Wos = ah(256 * 256), *W1s = ah(512 * 512),
           *W2s = ah(512 * 256);
  _Float16 *Wva = ah(256 * 256), *Woa = ah(256 * 256), *W1a = ah(512 * 512),
           *W2a = ah(512 * 256);
  _Float16 *Wqkc = ah(256 * 256), *Wvc = ah(256 * 256), *Woc = ah(256 * 256),
           *W1c = ah(512 * 512), *W2c = ah(512 * 256);
  _Float16* Wcp = ah(512 * 256);

  // activations
  const size_t RD = (size_t)BN * DD;   // 4096*256
  _Float16* X = ah(RD);
  _Float16* QKV = ah((size_t)BN * 768);
  _Float16 *Qh = ah(RD), *Kh = ah(RD), *Vh = ah(RD);   // Vh is d-major
  _Float16* CTX = ah(RD);
  _Float16* CAT = ah((size_t)BN * D2);
  _Float16* G16 = ah((size_t)BN * D2);
  _Float16 *D0 = ah(RD), *D1 = ah(RD);
  _Float16 *A0 = ah(RD), *A1 = ah(RD);
  _Float16 *V0b = ah(RD), *V1b = ah(RD);               // d-major
  _Float16* Mctx = ah(RD);
  _Float16 *Aout0 = ah(RD), *Aout1 = ah(RD);
  _Float16 *Dc0 = ah(RD), *Dc1 = ah(RD);
  _Float16 *QK0 = ah(RD), *QK1 = ah(RD);
  _Float16 *Vc0 = ah(RD), *Vc1 = ah(RD);               // d-major
  float* F32A = af((size_t)BN * D2);
  float* CE = af(512);

  auto cvtT = [&](const float* s, _Float16* d, int K, int Nn) {
    cvt_f32_f16_t<<<(K * Nn + 255) / 256, 256, 0, stream>>>(s, d, K, Nn);
  };
  cvtT(F(8), Wqkv, 256, 768);  cvtT(F(10), Wos, 256, 256);
  cvtT(F(12), W1s, 512, 512);  cvtT(F(16), W2s, 512, 256);
  cvtT(F(18), Wva, 256, 256);  cvtT(F(20), Woa, 256, 256);
  cvtT(F(22), W1a, 512, 512);  cvtT(F(26), W2a, 512, 256);
  cvtT(F(28), Wqkc, 256, 256); cvtT(F(30), Wvc, 256, 256);
  cvtT(F(32), Woc, 256, 256);  cvtT(F(34), W1c, 512, 512);
  cvtT(F(38), W2c, 512, 256);  cvtT(F(40), Wcp, 512, 256);

  auto g16 = [&](const _Float16* A, const _Float16* W, const float* bias,
                 _Float16* C, int M, int Nn, int K, int ldc, int coff) {
    gemm_wmma<0><<<dim3(M / 64, Nn / 64), 128, 0, stream>>>(
        A, W, bias, nullptr, C, nullptr, M, Nn, K, ldc, coff);
  };
  auto g32 = [&](const _Float16* A, const _Float16* W, const float* bias,
                 float* C, int M, int Nn, int K) {
    gemm_wmma<1><<<dim3(M / 64, Nn / 64), 128, 0, stream>>>(
        A, W, bias, nullptr, nullptr, C, M, Nn, K, Nn, 0);
  };
  auto g16r = [&](const _Float16* A, const _Float16* W, const float* bias,
                  const _Float16* R, _Float16* C, int M, int Nn, int K) {
    gemm_wmma<2><<<dim3(M / 64, Nn / 64), 128, 0, stream>>>(
        A, W, bias, R, C, nullptr, M, Nn, K, Nn, 0);
  };
  auto g32r = [&](const _Float16* A, const _Float16* W, const float* bias,
                  const _Float16* R, float* C, int M, int Nn, int K) {
    gemm_wmma<3><<<dim3(M / 64, Nn / 64), 128, 0, stream>>>(
        A, W, bias, R, nullptr, C, M, Nn, K, Nn, 0);
  };
  auto g16t = [&](const _Float16* A, const _Float16* W, const float* bias,
                  _Float16* C, int M, int Nn, int K) {   // d-major V producer
    gemm_wmma<4><<<dim3(M / 64, Nn / 64), 128, 0, stream>>>(
        A, W, bias, nullptr, C, nullptr, M, Nn, K, Nn, 0);
  };

  // shared FFN tail: out = Xres + ffn(cat[Xres, Mctx_in @ Wo + bo])
  auto msg_ffn = [&](const _Float16* Xres, const _Float16* Mi,
                     const _Float16* Wo, const float* bo, const _Float16* W1,
                     const float* b1, const float* gn, const float* bgn,
                     const _Float16* W2, const float* b2, _Float16* Out16,
                     float* Out32) {
    g16(Mi, Wo, bo, CAT, BN, 256, 256, 512, 256);                // CAT[:,256:]
    copy16<<<4096, 256, 0, stream>>>(Xres, 256, CAT, 512, 256);  // CAT[:,:256]
    g32(CAT, W1, b1, F32A, BN, 512, 512);
    ln_gelu<<<BN / 8, 256, 0, stream>>>(F32A, gn, bgn, G16);
    if (Out16) g16r(G16, W2, b2, Xres, Out16, BN, 256, 512);
    else       g32r(G16, W2, b2, Xres, Out32, BN, 256, 512);
  };

  // ---- stage 1: self blocks ----------------------------------------------
  auto self_block = [&](const float* desc, const float* enc, _Float16* Dout) {
    cvt_f32_f16<<<(BN * DD + 255) / 256, 256, 0, stream>>>(desc, X, BN * DD);
    g16(X, Wqkv, F(9), QKV, BN, 768, 256, 768, 0);
    qkv_rope<<<4096, 256, 0, stream>>>(QKV, enc, Qh, Kh, Vh);
    flash_attn<true><<<dim3(NN / 64, HH, BB), 128, 0, stream>>>(
        Qh, Kh, Vh, CTX, 0.125f);  // HD^-0.5
    msg_ffn(X, CTX, Wos, F(11), W1s, F(13), F(14), F(15), W2s, F(17), Dout,
            nullptr);
  };
  self_block(desc0, enc0, D0);
  self_block(desc1, enc1, D1);

  // ---- stage 2: conditioning + adjacency block ---------------------------
  ce_kernel<<<2, 256, 0, stream>>>(embd, F(6), F(7), CE);
  add_ce_k<<<4096, 256, 0, stream>>>(D0, CE, A0);
  add_ce_k<<<4096, 256, 0, stream>>>(D1, CE, A1);
  g16t(A0, Wva, F(19), V0b, BN, 256, 256);   // d-major V
  g16t(A1, Wva, F(19), V1b, BN, 256, 256);
  // m0 = softmax(adj) @ v1 ; m1 = softmax(adj^T) @ v0
  flash_adj<false><<<dim3(NN / 64, DD / 64, BB), 128, 0, stream>>>(adj, V1b,
                                                                   Mctx);
  msg_ffn(A0, Mctx, Woa, F(21), W1a, F(23), F(24), F(25), W2a, F(27), Aout0,
          nullptr);
  flash_adj<true><<<dim3(NN / 64, DD / 64, BB), 128, 0, stream>>>(adj, V0b,
                                                                  Mctx);
  msg_ffn(A1, Mctx, Woa, F(21), W1a, F(23), F(24), F(25), W2a, F(27), Aout1,
          nullptr);

  // ---- stage 3: compact ---------------------------------------------------
  copy16<<<4096, 256, 0, stream>>>(D0, 256, CAT, 512, 256);
  copy16<<<4096, 256, 0, stream>>>(Aout0, 256, CAT + 256, 512, 256);
  g16(CAT, Wcp, F(41), Dc0, BN, 256, 512, 256, 0);
  copy16<<<4096, 256, 0, stream>>>(D1, 256, CAT, 512, 256);
  copy16<<<4096, 256, 0, stream>>>(Aout1, 256, CAT + 256, 512, 256);
  g16(CAT, Wcp, F(41), Dc1, BN, 256, 512, 256, 0);

  // ---- stage 4: cross block ----------------------------------------------
  g16(Dc0, Wqkc, F(29), QK0, BN, 256, 256, 256, 0);
  g16(Dc1, Wqkc, F(29), QK1, BN, 256, 256, 256, 0);
  g16t(Dc0, Wvc, F(31), Vc0, BN, 256, 256);  // d-major V
  g16t(Dc1, Wvc, F(31), Vc1, BN, 256, 256);

  float* out0 = (float*)d_out;
  float* out1 = out0 + RD;
  // per-side scale s=HD^-0.25 on both q and k folds into sim scale HD^-0.5
  flash_attn<false><<<dim3(NN / 64, HH, BB), 128, 0, stream>>>(QK0, QK1, Vc1,
                                                               CTX, 0.125f);
  msg_ffn(Dc0, CTX, Woc, F(33), W1c, F(35), F(36), F(37), W2c, F(39), nullptr,
          out0);
  flash_attn<false><<<dim3(NN / 64, HH, BB), 128, 0, stream>>>(QK1, QK0, Vc0,
                                                               CTX, 0.125f);
  msg_ffn(Dc1, CTX, Woc, F(33), W1c, F(35), F(36), F(37), W2c, F(39), nullptr,
          out1);
}
